// DMTetGeometry_61692910240026
// MI455X (gfx1250) — compile-verified, matrix operand order checked by reading, not run on hardware
//
#include <hip/hip_runtime.h>
#include <hip/hip_bf16.h>
#include <stdint.h>

// ============================================================================
// DMTet marching-tetrahedra on MI455X (gfx1250, wave32).
//
// Deterministic, atomic-free pipeline:
//   K1  per-vertex: tanh(sdf), occupancy byte, scaled tanh(dir)
//   K2  per-candidate-edge (a*7+t): crossing flag -> per-block counts (WMMA)
//   K3  single-block exclusive scan (x3: edges, 1-tri tets, 2-tri tets)
//   K4  edge emit: compact index + interpolated 6-ch vertex features
//   K5  per-tet: tetindex -> 1-tri / 2-tri flags -> per-block counts (WMMA x2)
//   K6  face emit: triangle table (async-staged into LDS) -> face indices
//
// Kuhn-grid property used: every tet edge is (a, a+delta), delta in {0,1}^3,
// so unique edges enumerate as a*7 + (t-1), t = 4di+2dj+dk, in exactly
// jnp.unique's sorted order. inv(R) == R^T for the rotation matrices.
// ============================================================================

typedef __attribute__((ext_vector_type(16))) _Float16 v16h;
typedef __attribute__((ext_vector_type(8)))  float    v8f;

#define TPB 256

// TRIANGLE_TABLE (16x6) followed by NUM_TRI (16) -- staged into LDS via the
// gfx1250 async global->LDS path in the face kernel.
__device__ const int g_tables[112] = {
    -1,-1,-1,-1,-1,-1,
     1, 0, 2,-1,-1,-1,
     4, 0, 3,-1,-1,-1,
     1, 4, 2, 1, 3, 4,
     3, 1, 5,-1,-1,-1,
     2, 3, 0, 2, 5, 3,
     1, 4, 0, 1, 5, 4,
     4, 2, 5,-1,-1,-1,
     4, 5, 2,-1,-1,-1,
     4, 1, 0, 4, 5, 1,
     3, 2, 0, 3, 5, 2,
     1, 3, 5,-1,-1,-1,
     4, 1, 2, 4, 3, 1,
     3, 0, 4,-1,-1,-1,
     2, 0, 1,-1,-1,-1,
    -1,-1,-1,-1,-1,-1,
     0, 1, 1, 2, 1, 2, 2, 1, 1, 2, 2, 1, 2, 1, 1, 0 };

// Corner bitmasks of the 4 vertices of each of the 6 Kuhn tets
// (perms (4,2,1),(4,1,2),(2,4,1),(2,1,4),(1,4,2),(1,2,4) -> nested masks).
__device__ const int g_tet_masks[6][4] = {
    {0,4,6,7},{0,4,5,7},{0,2,6,7},{0,2,3,7},{0,1,5,7},{0,1,3,7}};

// BASE_TET_EDGES pairs
__device__ const int g_pairA[6] = {0,0,0,1,1,2};
__device__ const int g_pairB[6] = {1,2,3,2,3,3};

// ---------------------------------------------------------------------------
// Wave-level population count via the matrix pipe.
// Each lane drops its 0/1 flag into one f16 A slot; B = all ones. Then
// sum(D over all lanes/VGPRs) = sum_{m,k} A[m,k] * sum_n B[k,n] = 16 * sum(A)
// -- correct independent of the exact A/B/D VGPR striping. Exact in f32
// (<= 512). WMMA is TRANS-class on CDNA5, co-executing with VALU work.
// Requires EXEC all-ones: callers must not early-return before this.
// ---------------------------------------------------------------------------
__device__ __forceinline__ int wave_sum_wmma(int flag) {
    v16h a, b;
#pragma unroll
    for (int i = 0; i < 16; ++i) { a[i] = (_Float16)0.0f; b[i] = (_Float16)1.0f; }
    a[0] = (_Float16)(float)flag;
    v8f c = {};
    c = __builtin_amdgcn_wmma_f32_16x16x32_f16(false, a, false, b,
                                               (short)0, c, false, false);
    float s = 0.0f;
#pragma unroll
    for (int i = 0; i < 8; ++i) s += c[i];
#pragma unroll
    for (int off = 16; off > 0; off >>= 1) s += __shfl_xor(s, off, 32);
    return (int)(s * 0.0625f + 0.5f);   // ΣD / 16
}

// ---------------------------------------------------------------------------
__global__ void k_zero(float* o, int n) {
    int i = blockIdx.x * TPB + threadIdx.x;
    if (i < n) o[i] = 0.0f;
}

// K1: per-vertex preprocessing ----------------------------------------------
__global__ void k_vertex(const float* __restrict__ sdf_n,
                         const float* __restrict__ dir2,
                         float* __restrict__ sdf_t,
                         float* __restrict__ dirs,
                         unsigned char* __restrict__ occ,
                         int N, float dscale) {
    int n = blockIdx.x * TPB + threadIdx.x;
    if (n >= N) return;
    float st = tanhf(sdf_n[n]);
    sdf_t[n] = st;
    occ[n] = st > 0.0f ? 1 : 0;
    dirs[2 * n + 0] = dscale * tanhf(dir2[2 * n + 0]);
    dirs[2 * n + 1] = dscale * tanhf(dir2[2 * n + 1]);
}

// K2: crossing-edge flags -> per-block counts (WMMA) ------------------------
__global__ void k_edge_count(const unsigned char* __restrict__ occ,
                             int* __restrict__ blkCnt, int NE, int Rp1) {
    int c = blockIdx.x * TPB + threadIdx.x;
    int flag = 0;
    if (c < NE) {
        int a = c / 7, t = c % 7 + 1;
        int di = (t >> 2) & 1, dj = (t >> 1) & 1, dk = t & 1;
        int k = a % Rp1, j = (a / Rp1) % Rp1, i = a / (Rp1 * Rp1);
        if (i + di < Rp1 && j + dj < Rp1 && k + dk < Rp1) {
            int b = a + di * Rp1 * Rp1 + dj * Rp1 + dk;
            flag = ((int)occ[a] + (int)occ[b]) == 1;
        }
    }
    int wsum = wave_sum_wmma(flag);
    __shared__ int s_w[8];
    if ((threadIdx.x & 31) == 0) s_w[threadIdx.x >> 5] = wsum;
    __syncthreads();
    if (threadIdx.x == 0) {
        int t = 0;
#pragma unroll
        for (int w = 0; w < 8; ++w) t += s_w[w];
        blkCnt[blockIdx.x] = t;
    }
}

// K3: single-block exclusive scan over per-block counts ---------------------
__global__ void k_scan(const int* __restrict__ cnt, int* __restrict__ base,
                       int n, int* __restrict__ total) {
    __shared__ int tmp[TPB];
    __shared__ int carry;
    if (threadIdx.x == 0) carry = 0;
    __syncthreads();
    for (int s = 0; s < n; s += TPB) {
        int i = s + threadIdx.x;
        int v = (i < n) ? cnt[i] : 0;
        tmp[threadIdx.x] = v;
        __syncthreads();
        for (int off = 1; off < TPB; off <<= 1) {
            int t = (threadIdx.x >= off) ? tmp[threadIdx.x - off] : 0;
            __syncthreads();
            tmp[threadIdx.x] += t;
            __syncthreads();
        }
        int incl = tmp[threadIdx.x];
        if (i < n) base[i] = carry + incl - v;
        __syncthreads();
        if (threadIdx.x == TPB - 1) carry += incl;
        __syncthreads();
    }
    if (threadIdx.x == 0 && total) *total = carry;
}

// K4: edge compaction + interpolated vertex features ------------------------
__global__ void k_edge_emit(const unsigned char* __restrict__ occ,
                            const float* __restrict__ sdf_t,
                            const float* __restrict__ dirs,
                            const float* __restrict__ feats,
                            const float* __restrict__ rot,
                            const float* __restrict__ scl,
                            const int* __restrict__ blkBase,
                            int* __restrict__ edge_index,
                            float* __restrict__ out_vf,
                            int NE, int Rp1) {
    int c = blockIdx.x * TPB + threadIdx.x;
    int a = 0, b = 0, flag = 0;
    if (c < NE) {
        a = c / 7;
        int t = c % 7 + 1;
        int di = (t >> 2) & 1, dj = (t >> 1) & 1, dk = t & 1;
        int k = a % Rp1, j = (a / Rp1) % Rp1, i = a / (Rp1 * Rp1);
        if (i + di < Rp1 && j + dj < Rp1 && k + dk < Rp1) {
            b = a + di * Rp1 * Rp1 + dj * Rp1 + dk;
            flag = ((int)occ[a] + (int)occ[b]) == 1;
        }
    }
    unsigned mask = __builtin_amdgcn_ballot_w32(flag != 0);
    int lane = threadIdx.x & 31, wave = threadIdx.x >> 5;
    __shared__ int s_w[8];
    if (lane == 0) s_w[wave] = __popc(mask);
    __syncthreads();
    int woff = 0;
    for (int w = 0; w < wave; ++w) woff += s_w[w];
    int rank = __popc(mask & ((1u << lane) - 1u));
    int idx = flag ? (blkBase[blockIdx.x] + woff + rank) : -1;
    if (c < NE) edge_index[c] = idx;

    if (flag) {
        float sa = sdf_t[a], sb = sdf_t[b];
        float inv = 1.0f / (sa - sb);
        float w0 = -sb * inv, w1 = sa * inv;   // reference: s2[::-1]/sum
        const float* Ra = rot + (size_t)a * 9;
        const float* Rb = rot + (size_t)b * 9;
        float axa0 = dirs[2 * a], axa1 = dirs[2 * a + 1];
        float axb0 = dirs[2 * b], axb1 = dirs[2 * b + 1];
        float sca = scl[a], scb = scl[b];
        float ea[6], eb[6];
#pragma unroll
        for (int j2 = 0; j2 < 3; ++j2) {   // rotated = (axis . R), i.e. R^T axis
            ea[j2] = (axa0 * Ra[0 * 3 + j2] + axa1 * Ra[1 * 3 + j2] + Ra[2 * 3 + j2]) * sca;
            eb[j2] = (axb0 * Rb[0 * 3 + j2] + axb1 * Rb[1 * 3 + j2] + Rb[2 * 3 + j2]) * scb;
        }
#pragma unroll
        for (int cc = 3; cc < 6; ++cc) {
            ea[cc] = feats[(size_t)a * 6 + cc];
            eb[cc] = feats[(size_t)b * 6 + cc];
        }
#pragma unroll
        for (int cc = 0; cc < 6; ++cc)
            out_vf[(size_t)idx * 6 + cc] = w0 * ea[cc] + w1 * eb[cc];
    }
}

// Shared tet decode ----------------------------------------------------------
__device__ __forceinline__ int tet_index(const unsigned char* occ, int f,
                                         int R, int tv[4]) {
    int R3 = R * R * R;
    int p = f / R3, cidx = f % R3;
    int ck = cidx % R, cj = (cidx / R) % R, ci = cidx / (R * R);
    int ti = 0;
#pragma unroll
    for (int v = 0; v < 4; ++v) {
        int m = g_tet_masks[p][v];
        int vi = ci + ((m >> 2) & 1), vj = cj + ((m >> 1) & 1), vk = ck + (m & 1);
        int vid = (vi * (R + 1) + vj) * (R + 1) + vk;
        tv[v] = vid;
        ti |= ((int)occ[vid]) << v;
    }
    return ti;
}

// K5: per-tet face-count flags -> per-block counts (WMMA x2) ----------------
__global__ void k_tet_count(const unsigned char* __restrict__ occ,
                            int* __restrict__ cnt1, int* __restrict__ cnt2,
                            int R, int F) {
    int f = blockIdx.x * TPB + threadIdx.x;
    int f1 = 0, f2 = 0;
    if (f < F) {
        int tv[4];
        int ti = tet_index(occ, f, R, tv);
        int pc = __popc((unsigned)ti);
        f1 = (pc == 1) | (pc == 3);
        f2 = (pc == 2);
    }
    int w1 = wave_sum_wmma(f1);
    int w2 = wave_sum_wmma(f2);
    __shared__ int s_a[8], s_b[8];
    if ((threadIdx.x & 31) == 0) {
        s_a[threadIdx.x >> 5] = w1;
        s_b[threadIdx.x >> 5] = w2;
    }
    __syncthreads();
    if (threadIdx.x == 0) {
        int t1 = 0, t2 = 0;
#pragma unroll
        for (int w = 0; w < 8; ++w) { t1 += s_a[w]; t2 += s_b[w]; }
        cnt1[blockIdx.x] = t1;
        cnt2[blockIdx.x] = t2;
    }
}

// K6: face emission. Triangle tables async-staged into LDS ------------------
__global__ void k_face_emit(const unsigned char* __restrict__ occ,
                            const int* __restrict__ edge_index,
                            const int* __restrict__ base1,
                            const int* __restrict__ base2,
                            const int* __restrict__ hdr,
                            float* __restrict__ d_out,
                            int R, int F) {
    __shared__ int s_tab[112];
    if (threadIdx.x < 112) {
        const int* gp = &g_tables[threadIdx.x];
        unsigned lds = (unsigned)(size_t)&s_tab[threadIdx.x];
        // gfx1250 async global->LDS copy (ASYNCcnt-tracked), ISA §10 / §15.18.3
        asm volatile("global_load_async_to_lds_b32 %0, %1, off"
                     :: "v"(lds), "v"(gp) : "memory");
    }
    asm volatile("s_wait_asynccnt 0x0" ::: "memory");
    __syncthreads();

    int f = blockIdx.x * TPB + threadIdx.x;
    int tv[4] = {0, 0, 0, 0};
    int ti = 0, nt = 0;
    if (f < F) {
        ti = tet_index(occ, f, R, tv);
        int pc = __popc((unsigned)ti);
        nt = (pc == 2) ? 2 : ((pc == 1 || pc == 3) ? 1 : 0);
    }
    unsigned m1 = __builtin_amdgcn_ballot_w32(nt == 1);
    unsigned m2 = __builtin_amdgcn_ballot_w32(nt == 2);
    int lane = threadIdx.x & 31, wave = threadIdx.x >> 5;
    __shared__ int s_w1[8], s_w2[8];
    if (lane == 0) { s_w1[wave] = __popc(m1); s_w2[wave] = __popc(m2); }
    __syncthreads();
    int woff1 = 0, woff2 = 0;
    for (int w = 0; w < wave; ++w) { woff1 += s_w1[w]; woff2 += s_w2[w]; }
    int r1 = base1[blockIdx.x] + woff1 + __popc(m1 & ((1u << lane) - 1u));
    int r2 = base2[blockIdx.x] + woff2 + __popc(m2 & ((1u << lane) - 1u));

    if (nt) {
        int NV  = hdr[0];
        int NF1 = hdr[1];
        size_t fbase = (size_t)NV * 6;                 // faces follow vert_feats
        size_t row = (nt == 1) ? (size_t)r1 : (size_t)NF1 + 2u * (size_t)r2;
        const int* tt = &s_tab[ti * 6];
        int S2 = (R + 1) * (R + 1);
        for (int tri = 0; tri < nt; ++tri) {
            for (int e = 0; e < 3; ++e) {
                int slot = tt[tri * 3 + e];
                int va = tv[g_pairA[slot]], vb = tv[g_pairB[slot]];
                int delta = vb - va;                   // masks nested => va < vb
                int di = delta >= S2;       delta -= di * S2;
                int dj = delta >= (R + 1);  delta -= dj * (R + 1);
                int dk = delta;
                int t = di * 4 + dj * 2 + dk;
                int idx = edge_index[va * 7 + t - 1];
                d_out[fbase + (row + tri) * 3 + e] = (float)idx;
            }
        }
    }
}

// ===========================================================================
extern "C" void kernel_launch(void* const* d_in, const int* in_sizes, int n_in,
                              void* d_out, int out_size, void* d_ws, size_t ws_size,
                              hipStream_t stream) {
    const float* feats = (const float*)d_in[0];
    const float* sdf   = (const float*)d_in[1];
    const float* dir2  = (const float*)d_in[2];
    const float* rot   = (const float*)d_in[3];
    const float* scl   = (const float*)d_in[4];
    (void)n_in; (void)ws_size;

    int N = in_sizes[1];                 // (R+1)^3 vertices
    int Rp1 = 1;
    while ((long long)Rp1 * Rp1 * Rp1 < (long long)N) ++Rp1;
    int R  = Rp1 - 1;
    int NE = N * 7;
    int F  = 6 * R * R * R;
    int nbE = (NE + TPB - 1) / TPB;
    int nbT = (F + TPB - 1) / TPB;

    // workspace carving (16B aligned)
    uint8_t* w = (uint8_t*)d_ws;
    size_t off = 0;
    auto carve = [&](size_t bytes) -> void* {
        void* p = w + off;
        off += (bytes + 15) & ~(size_t)15;
        return p;
    };
    int*           hdr        = (int*)carve(64);            // [0]=NV [1]=NF1 [2]=NF2
    unsigned char* occ        = (unsigned char*)carve((size_t)N);
    float*         sdf_t      = (float*)carve((size_t)N * 4);
    float*         dirs       = (float*)carve((size_t)N * 8);
    int*           edge_index = (int*)carve((size_t)NE * 4);
    int*           cntE       = (int*)carve((size_t)nbE * 4);
    int*           baseE      = (int*)carve((size_t)nbE * 4);
    int*           cnt1       = (int*)carve((size_t)nbT * 4);
    int*           cnt2       = (int*)carve((size_t)nbT * 4);
    int*           base1      = (int*)carve((size_t)nbT * 4);
    int*           base2      = (int*)carve((size_t)nbT * 4);

    float* out = (float*)d_out;
    if (out_size > 0)
        k_zero<<<(out_size + TPB - 1) / TPB, TPB, 0, stream>>>(out, out_size);

    k_vertex<<<(N + TPB - 1) / TPB, TPB, 0, stream>>>(
        sdf, dir2, sdf_t, dirs, occ, N, 4.0f / (float)R);

    k_edge_count<<<nbE, TPB, 0, stream>>>(occ, cntE, NE, Rp1);
    k_scan<<<1, TPB, 0, stream>>>(cntE, baseE, nbE, hdr + 0);
    k_edge_emit<<<nbE, TPB, 0, stream>>>(occ, sdf_t, dirs, feats, rot, scl,
                                         baseE, edge_index, out, NE, Rp1);

    k_tet_count<<<nbT, TPB, 0, stream>>>(occ, cnt1, cnt2, R, F);
    k_scan<<<1, TPB, 0, stream>>>(cnt1, base1, nbT, hdr + 1);
    k_scan<<<1, TPB, 0, stream>>>(cnt2, base2, nbT, hdr + 2);
    k_face_emit<<<nbT, TPB, 0, stream>>>(occ, edge_index, base1, base2, hdr,
                                         out, R, F);
}
